// acf_Module_7851200217317
// MI455X (gfx1250) — matrix operand order repaired and checked by reading, not compiled, not run on hardware
//
#include <hip/hip_runtime.h>

// ---------------------------------------------------------------------------
// Types
// ---------------------------------------------------------------------------
typedef __attribute__((ext_vector_type(16))) __bf16 v16bf;
typedef __attribute__((ext_vector_type(8)))  float  v8f;
typedef __attribute__((ext_vector_type(4)))  float  v4f;

union BfFrag {          // 16 bf16 = 32 bytes = 8 VGPRs (A or B operand)
    v16bf v;
    uint4 u[2];
};

// round-half-up: 1 VALU, result in high 16 bits -> ds_store_b16_d16_hi
__device__ __forceinline__ unsigned short f2bf(float f) {
    return (unsigned short)((__float_as_uint(f) + 0x8000u) >> 16);
}

__device__ __forceinline__ v8f v8f_zero() {
    v8f z = {0.f, 0.f, 0.f, 0.f, 0.f, 0.f, 0.f, 0.f};
    return z;
}

// Problem constants
#define B_    8
#define CIN_  512
#define C_    256
#define N_    19
#define S_    16384          // H*W
#define S_TILE 64
#define NSB   (S_ / S_TILE)  // 256 s-blocks
#define XPAD  40             // halves stride of lds_x  (80B, 16B-mult)
#define FPAD  72             // halves stride of lds_feat (144B, 16B-mult)
#define XBUF  (S_TILE * XPAD) // one lds_x buffer in halves (5120B)

// Workspace layout (bytes)
#define W1S_OFF  0                      // bf16 [256][512]            = 262144
#define BIAS_OFF (W1S_OFF + 262144)     // f32 [256]                  = 1024
#define EP_OFF   (BIAS_OFF + 1024)      // f32 [8][256][19][256]      = 39845888
#define ATTN_OFF (EP_OFF + 39845888)    // f32 [8][19][256]           = 155648
#define M_OFF    (ATTN_OFF + 155648)    // f32 [8][256][19]           = 155648

// ---------------------------------------------------------------------------
// k0: fold BN scale into conv1 weights (-> bf16), compute BN bias
// ---------------------------------------------------------------------------
__global__ __launch_bounds__(256) void k0_prep(
    const float* __restrict__ w1, const float* __restrict__ gamma,
    const float* __restrict__ beta, const float* __restrict__ mean,
    const float* __restrict__ var,
    unsigned short* __restrict__ w1s, float* __restrict__ biasArr)
{
    int g = blockIdx.x * 256 + threadIdx.x;
    if (g < C_ * CIN_) {
        int c = g >> 9;                       // / 512
        float sc = gamma[c] * rsqrtf(var[c] + 1e-5f);
        w1s[g] = f2bf(w1[g] * sc);
    }
    if (g < C_) {
        float sc = gamma[g] * rsqrtf(var[g] + 1e-5f);
        biasArr[g] = beta[g] - mean[g] * sc;
    }
}

// ---------------------------------------------------------------------------
// k1 (fused): conv1 (bf16 WMMA, fp32 acc) + BN bias + ReLU  ->  LDS feat tile
//             -> energy partials  ep[b][sb][n][c] = q * feat  (bf16 WMMA)
// Grid: (256 s-blocks, 8 batches), 256 threads = 8 waves.
// Wave w owns c-rows [w*32, w*32+32), full 64-wide s tile.
// Double-buffered X staging: one barrier per K-chunk.
// ---------------------------------------------------------------------------
__global__ __launch_bounds__(256) void k1_conv_energy(
    const float* __restrict__ x,              // feat_ffm [8][512][16384]
    const float* __restrict__ q,              // coarse_x [8][19][16384]
    const unsigned short* __restrict__ w1s,   // bf16 [256][512]
    const float* __restrict__ biasArr,        // [256]
    float* __restrict__ ep)                   // [8][256][19][256]
{
    __shared__ alignas(16) unsigned short lds_x[2 * XBUF];        // 10240 B
    __shared__ alignas(16) unsigned short lds_feat[C_ * FPAD];    // 36864 B
    __shared__ alignas(16) unsigned short lds_q[32 * S_TILE];     //  4096 B
    __shared__ float lds_bias[C_];

    const int sb   = blockIdx.x;
    const int b    = blockIdx.y;
    const int tid  = threadIdx.x;
    const int wave = tid >> 5;
    const int lane = tid & 31;
    const int l15  = lane & 15;
    const int lhi  = lane >> 4;              // 0 / 1 (half-wave)
    const int s_base = sb * S_TILE;

    // stage q tile as bf16, rows n>=19 zero-padded to 32
    for (int idx = tid; idx < 32 * S_TILE; idx += 256) {
        int n = idx / S_TILE, s = idx % S_TILE;
        float v = (n < N_) ? q[((size_t)(b * N_ + n)) * S_ + s_base + s] : 0.f;
        lds_q[n * S_TILE + s] = f2bf(v);
    }
    lds_bias[tid] = biasArr[tid];

    v8f acc[2][4];
    for (int i = 0; i < 2; ++i)
        for (int j = 0; j < 4; ++j) acc[i][j] = v8f_zero();

    const int cwave = wave * 32;

    // hoisted staging geometry (loop-invariant)
    const int r0 = tid >> 4;                 // 0..15 (k row within half-chunk)
    const int sc = (tid & 15) * 4;           // s offset, float4 aligned
    const float* xp = x + ((size_t)(b * CIN_) + r0) * S_ + s_base + sc;
    unsigned short* const xst = lds_x + sc * XPAD + r0;

    // hoisted A-fragment bases (bf16 weights, L2-resident)
    const unsigned short* wp[2];
    for (int ci = 0; ci < 2; ++ci)
        wp[ci] = w1s + (size_t)(cwave + ci * 16 + l15) * CIN_ + lhi * 8;

    // ---- conv1 GEMM: K = 512 in 16 chunks of 32, double-buffered ----
    for (int kc = 0; kc < 16; ++kc) {
        const int bsel = (kc & 1) * XBUF;
        // stage X chunk [32 k][64 s] -> lds_x transposed [s][k] (bf16)
        {
            unsigned short* dst = xst + bsel;
            const v4f x0 = __builtin_nontemporal_load(
                reinterpret_cast<const v4f*>(xp));
            const v4f x1 = __builtin_nontemporal_load(
                reinterpret_cast<const v4f*>(xp + 16 * S_));  // +1MB imm24
            dst[0 * XPAD]      = f2bf(x0.x);
            dst[1 * XPAD]      = f2bf(x0.y);
            dst[2 * XPAD]      = f2bf(x0.z);
            dst[3 * XPAD]      = f2bf(x0.w);
            dst[0 * XPAD + 16] = f2bf(x1.x);
            dst[1 * XPAD + 16] = f2bf(x1.y);
            dst[2 * XPAD + 16] = f2bf(x1.z);
            dst[3 * XPAD + 16] = f2bf(x1.w);
            xp += 32 * S_;
        }
        __syncthreads();

        // A fragments (global bf16 weights)
        BfFrag afr[2];
        for (int ci = 0; ci < 2; ++ci) {
            afr[ci].u[0] = *reinterpret_cast<const uint4*>(wp[ci]);
            afr[ci].u[1] = *reinterpret_cast<const uint4*>(wp[ci] + 16);
            wp[ci] += 32;
        }
        // B fragments from LDS, 8 WMMAs per wave per chunk
        for (int sj = 0; sj < 4; ++sj) {
            BfFrag bfr;
            const unsigned short* src =
                &lds_x[bsel + (sj * 16 + l15) * XPAD + lhi * 8];
            bfr.u[0] = *reinterpret_cast<const uint4*>(src);
            bfr.u[1] = *reinterpret_cast<const uint4*>(src + 16);
            for (int ci = 0; ci < 2; ++ci)
                acc[ci][sj] = __builtin_amdgcn_wmma_f32_16x16x32_bf16(
                    false, afr[ci].v, false, bfr.v, (short)0, acc[ci][sj],
                    false, false);
        }
    }

    // ---- epilogue: + bias, ReLU, write bf16 feat tile to LDS ----
    for (int ci = 0; ci < 2; ++ci)
        for (int sj = 0; sj < 4; ++sj)
            for (int r = 0; r < 8; ++r) {
                int c = cwave + ci * 16 + r + 8 * lhi;
                int s = sj * 16 + l15;
                float v = acc[ci][sj][r] + lds_bias[c];
                v = v > 0.f ? v : 0.f;
                lds_feat[c * FPAD + s] = f2bf(v);
            }
    __syncthreads();

    // ---- energy: e[n,c] = sum_s q[n,s]*feat[c,s]  (A=q 16x32, B=feat^T 32x16)
    v8f eacc[2][2];
    for (int t = 0; t < 2; ++t)
        for (int cj = 0; cj < 2; ++cj) eacc[t][cj] = v8f_zero();

    for (int kc2 = 0; kc2 < 2; ++kc2) {          // s in two 32-chunks
        BfFrag qa[2];
        for (int t = 0; t < 2; ++t) {
            const unsigned short* src =
                &lds_q[(t * 16 + l15) * S_TILE + kc2 * 32 + lhi * 8];
            qa[t].u[0] = *reinterpret_cast<const uint4*>(src);
            qa[t].u[1] = *reinterpret_cast<const uint4*>(src + 16);
        }
        for (int cj = 0; cj < 2; ++cj) {
            BfFrag fb;
            const unsigned short* src =
                &lds_feat[(cwave + cj * 16 + l15) * FPAD + kc2 * 32 + lhi * 8];
            fb.u[0] = *reinterpret_cast<const uint4*>(src);
            fb.u[1] = *reinterpret_cast<const uint4*>(src + 16);
            for (int t = 0; t < 2; ++t)
                eacc[t][cj] = __builtin_amdgcn_wmma_f32_16x16x32_bf16(
                    false, qa[t].v, false, fb.v, (short)0, eacc[t][cj],
                    false, false);
        }
    }

    // deterministic partials (no atomics)
    for (int t = 0; t < 2; ++t)
        for (int cj = 0; cj < 2; ++cj)
            for (int r = 0; r < 8; ++r) {
                int n = t * 16 + r + 8 * lhi;
                if (n < N_) {
                    int c = cwave + cj * 16 + l15;
                    ep[(((size_t)(b * NSB + sb)) * N_ + n) * C_ + c] = eacc[t][cj][r];
                }
            }
}

// ---------------------------------------------------------------------------
// k2: reduce energy partials over 256 s-blocks, softmax over C=256
// Grid: (19, 8), 256 threads (one per c)
// ---------------------------------------------------------------------------
__global__ __launch_bounds__(256) void k2_softmax(
    const float* __restrict__ ep, float* __restrict__ attn)
{
    __shared__ float red[256];
    const int n = blockIdx.x, b = blockIdx.y, c = threadIdx.x;

    float acc = 0.f;
    const float* p = ep + (((size_t)(b * NSB)) * N_ + n) * C_ + c;
    for (int sb = 0; sb < NSB; ++sb) acc += p[(size_t)sb * N_ * C_];

    red[c] = acc;
    __syncthreads();
    for (int off = 128; off > 0; off >>= 1) {
        if (c < off) red[c] = fmaxf(red[c], red[c + off]);
        __syncthreads();
    }
    float mx = red[0];
    __syncthreads();
    float e = __expf(acc - mx);
    red[c] = e;
    __syncthreads();
    for (int off = 128; off > 0; off >>= 1) {
        if (c < off) red[c] += red[c + off];
        __syncthreads();
    }
    attn[(((size_t)b) * N_ + n) * C_ + c] = e / red[0];
}

// ---------------------------------------------------------------------------
// k3: M[b][o][n] = sum_c conv2_w[o][c] * attn[b][n][c]   (tiny GEMM)
// Grid: 8 blocks (b), 256 threads (o)
// ---------------------------------------------------------------------------
__global__ __launch_bounds__(256) void k3_M(
    const float* __restrict__ attn, const float* __restrict__ w2,
    float* __restrict__ Mg)
{
    __shared__ float a_lds[N_ * C_];
    const int b = blockIdx.x, o = threadIdx.x;
    for (int idx = o; idx < N_ * C_; idx += 256)
        a_lds[idx] = attn[(size_t)b * N_ * C_ + idx];
    __syncthreads();

    float accn[N_];
    for (int n = 0; n < N_; ++n) accn[n] = 0.f;
    for (int c = 0; c < C_; ++c) {
        float wv = w2[(size_t)o * C_ + c];
        for (int n = 0; n < N_; ++n) accn[n] += wv * a_lds[n * C_ + c];
    }
    for (int n = 0; n < N_; ++n)
        Mg[(((size_t)b) * C_ + o) * N_ + n] = accn[n];
}

// ---------------------------------------------------------------------------
// k4: out[b][o][s] = sum_n M[b][o][n]*q[b][n][s] + b2[o]   (WMMA, K=32 padded)
// Grid: (128 s-blocks of 128, 8 batches), 256 threads = 8 waves.
// ---------------------------------------------------------------------------
__global__ __launch_bounds__(256) void k4_out(
    const float* __restrict__ q, const float* __restrict__ Mg,
    const float* __restrict__ b2, float* __restrict__ out)
{
    __shared__ alignas(16) unsigned short lds_m[C_ * 32];    // 16 KB
    __shared__ alignas(16) unsigned short lds_qt[128 * 32];  //  8 KB
    __shared__ float lds_b2[C_];

    const int sblk = blockIdx.x;
    const int b    = blockIdx.y;
    const int tid  = threadIdx.x;
    const int wave = tid >> 5;
    const int lane = tid & 31;
    const int l15  = lane & 15;
    const int lhi  = lane >> 4;
    const int s_base = sblk * 128;

    // M[b] -> bf16 [o][npad32]
    for (int idx = tid; idx < C_ * 32; idx += 256) {
        int o = idx >> 5, n = idx & 31;
        float v = (n < N_) ? Mg[(((size_t)b) * C_ + o) * N_ + n] : 0.f;
        lds_m[idx] = f2bf(v);
    }
    // q tile -> bf16 transposed [s][npad32] (coalesced reads over s)
    for (int idx = tid; idx < N_ * 128; idx += 256) {
        int n = idx / 128, s = idx % 128;
        lds_qt[s * 32 + n] =
            f2bf(q[(((size_t)b) * N_ + n) * S_ + s_base + s]);
    }
    for (int idx = tid; idx < (32 - N_) * 128; idx += 256) {
        int n = N_ + idx / 128, s = idx % 128;
        lds_qt[s * 32 + n] = 0;
    }
    lds_b2[tid] = b2[tid];
    __syncthreads();

    const int ow = wave * 32;
    BfFrag am[2];
    for (int ci = 0; ci < 2; ++ci) {
        const unsigned short* src = &lds_m[(ow + ci * 16 + l15) * 32 + lhi * 8];
        am[ci].u[0] = *reinterpret_cast<const uint4*>(src);
        am[ci].u[1] = *reinterpret_cast<const uint4*>(src + 16);
    }
    for (int sj = 0; sj < 8; ++sj) {
        BfFrag bq;
        const unsigned short* src = &lds_qt[(sj * 16 + l15) * 32 + lhi * 8];
        bq.u[0] = *reinterpret_cast<const uint4*>(src);
        bq.u[1] = *reinterpret_cast<const uint4*>(src + 16);
        for (int ci = 0; ci < 2; ++ci) {
            v8f acc = v8f_zero();
            acc = __builtin_amdgcn_wmma_f32_16x16x32_bf16(
                false, am[ci].v, false, bq.v, (short)0, acc, false, false);
            for (int r = 0; r < 8; ++r) {
                int o = ow + ci * 16 + r + 8 * lhi;
                int s = s_base + sj * 16 + l15;
                out[(((size_t)b) * C_ + o) * S_ + s] = acc[r] + lds_b2[o];
            }
        }
    }
}

// ---------------------------------------------------------------------------
// launch
// ---------------------------------------------------------------------------
extern "C" void kernel_launch(void* const* d_in, const int* in_sizes, int n_in,
                              void* d_out, int out_size, void* d_ws, size_t ws_size,
                              hipStream_t stream) {
    (void)in_sizes; (void)n_in; (void)out_size; (void)ws_size;
    const float* feat_ffm = (const float*)d_in[0];
    const float* coarse_x = (const float*)d_in[1];
    const float* conv1_w  = (const float*)d_in[2];
    const float* bn_gamma = (const float*)d_in[3];
    const float* bn_beta  = (const float*)d_in[4];
    const float* bn_mean  = (const float*)d_in[5];
    const float* bn_var   = (const float*)d_in[6];
    const float* conv2_w  = (const float*)d_in[7];
    const float* conv2_b  = (const float*)d_in[8];
    float* out = (float*)d_out;

    char* ws = (char*)d_ws;
    unsigned short* w1s  = (unsigned short*)(ws + W1S_OFF);
    float* biasArr       = (float*)(ws + BIAS_OFF);
    float* ep            = (float*)(ws + EP_OFF);
    float* attn          = (float*)(ws + ATTN_OFF);
    float* Mg            = (float*)(ws + M_OFF);

    k0_prep<<<dim3(512), 256, 0, stream>>>(conv1_w, bn_gamma, bn_beta,
                                           bn_mean, bn_var, w1s, biasArr);
    k1_conv_energy<<<dim3(NSB, B_), 256, 0, stream>>>(feat_ffm, coarse_x,
                                                      w1s, biasArr, ep);
    k2_softmax<<<dim3(N_, B_), 256, 0, stream>>>(ep, attn);
    k3_M<<<dim3(B_), 256, 0, stream>>>(attn, conv2_w, Mg);
    k4_out<<<dim3(S_ / 128, B_), 256, 0, stream>>>(coarse_x, Mg, conv2_b, out);
}
